// MultiHeadAttention_4827543241365
// MI455X (gfx1250) — matrix operand order repaired and checked by reading, not compiled, not run on hardware
//
#include <hip/hip_runtime.h>
#include <hip/hip_bf16.h>
#include <math.h>

// MI455X / gfx1250 multi-head attention, wave32 + v_wmma_f32_16x16x32_f16.
// Pipeline: 3x projection GEMM (f32 in -> f16; Q/K in [B,H,S,HD], V transposed
//           to [B,H,HD,S] so attention can DMA it straight into LDS)
//           1x flash-attention (async global->LDS tile staging, online softmax,
//           causal, WMMA QK^T and PV)
//           1x output GEMM (f32 out).
// Workspace: 16MB q + 16MB k + 16MB v^T (f16) + 32MB attn-out (f32) = 80MB.

typedef __attribute__((ext_vector_type(16))) _Float16 v16h;
typedef __attribute__((ext_vector_type(8)))  _Float16 v8h;
typedef __attribute__((ext_vector_type(8)))  float    v8f;
typedef __attribute__((__vector_size__(16))) int      vi4;   // b128 payload type

#define D_MODEL   1024
#define N_HEADS   16
#define HEAD_DIM  64
#define BATCH     4
#define SEQ       2048
#define M_ROWS    (BATCH * SEQ)   // 8192

// -------- CDNA5 async global->LDS copy (ASYNCcnt path), with fallback -------

#if defined(__has_builtin)
#if __has_builtin(__builtin_amdgcn_global_load_async_to_lds_b128)
#define HAVE_ASYNC_LDS 1
#endif
#endif
#ifndef HAVE_ASYNC_LDS
#define HAVE_ASYNC_LDS 0
#endif

#define AS1 __attribute__((address_space(1)))
#define AS3 __attribute__((address_space(3)))

#if HAVE_ASYNC_LDS
__device__ __forceinline__ void async_copy16(const _Float16* g, _Float16* l) {
    __builtin_amdgcn_global_load_async_to_lds_b128(
        (AS1 vi4*)(g), (AS3 vi4*)(l), /*offset=*/0, /*cpol=*/0);
}
__device__ __forceinline__ void wait_async() {
#if __has_builtin(__builtin_amdgcn_s_wait_asynccnt)
    __builtin_amdgcn_s_wait_asynccnt(0);
#else
    asm volatile("s_wait_asynccnt 0x0" ::: "memory");
#endif
}
#endif

// ---------------- fragment helpers (ISA 7.12.2 layouts) ----------------

__device__ __forceinline__ v16h cat8(v8h lo, v8h hi) {
    return __builtin_shufflevector(lo, hi, 0,1,2,3,4,5,6,7,8,9,10,11,12,13,14,15);
}

// Load 16 halves: chunk0 at p, chunk1 at p+gap (gap=16 for both A- and B-frags: AMD table k order)
__device__ __forceinline__ v16h ldfrag(const _Float16* p, int gap) {
    v8h lo = *(const v8h*)p;
    v8h hi = *(const v8h*)(p + gap);
    return cat8(lo, hi);
}

__device__ __forceinline__ v8f wmma16(v16h a, v16h b, v8f c) {
    return __builtin_amdgcn_wmma_f32_16x16x32_f16(false, a, false, b, (short)0, c,
                                                  false, false);
}

// ---------------- GEMM: C[M,N] = A[M,K] @ B[K,N] + bias ----------------
// A,B are f32 row-major in global; converted to f16 while staging into LDS.
// MODE 0: out = f32 [M,N] (final projection)
// MODE 1: out = f16 in [B, H, S, HD] head-split layout, scaled by `scale`
// MODE 2: out = f16 in [B, H, HD, S] (transposed heads, for V)
// Block: 256 thr (8 waves), tile 128x128, BK=32. Wave grid 4(M) x 2(N);
// each wave: 2x4 tiles of 16x16 -> 8 v_wmma per k-step.

#define GBM   128
#define GBN   128
#define GBK   32
#define APAD  40   // halves per A-tile row in LDS (80B: 16B-aligned, bank-spread)
#define BPAD  40   // halves per B-tile column in LDS (stored transposed)

template<int MODE>
__global__ __launch_bounds__(256)
void mha_gemm_kernel(const float* __restrict__ A, const float* __restrict__ Bw,
                     const float* __restrict__ bias, void* __restrict__ outp,
                     float scale)
{
    __shared__ _Float16 Asm[GBM * APAD];
    __shared__ _Float16 Bsm[GBN * BPAD];   // [n][k] (transposed)

    const int tid  = threadIdx.x;
    const int lane = tid & 31;
    const int wid  = tid >> 5;
    const int wm   = wid & 3;        // wave pos along M (0..3)
    const int wn   = wid >> 2;       // wave pos along N (0..1)
    const int hw   = lane >> 4;      // half-wave
    const int ln   = lane & 15;
    const int m0   = blockIdx.y * GBM;
    const int n0   = blockIdx.x * GBN;

    v8f acc[2][4];
#pragma unroll
    for (int i = 0; i < 2; ++i)
#pragma unroll
        for (int j = 0; j < 4; ++j)
#pragma unroll
            for (int r = 0; r < 8; ++r) acc[i][j][r] = 0.0f;

    // global->LDS staging assignments
    const int arow = tid >> 1;            // 0..127
    const int acol = (tid & 1) * 16;      // 0 or 16
    const int bk   = tid >> 3;            // 0..31
    const int bn   = (tid & 7) * 16;      // 0..112

    for (int k0 = 0; k0 < D_MODEL; k0 += GBK) {
        // ---- stage A tile (convert f32 -> f16) ----
        {
            const float4* ga = (const float4*)(A + (size_t)(m0 + arow) * D_MODEL
                                               + k0 + acol);
            union { _Float16 h[16]; v8h v[2]; } u;
#pragma unroll
            for (int i = 0; i < 4; ++i) {
                float4 f = ga[i];
                u.h[4*i+0] = (_Float16)f.x; u.h[4*i+1] = (_Float16)f.y;
                u.h[4*i+2] = (_Float16)f.z; u.h[4*i+3] = (_Float16)f.w;
            }
            _Float16* dst = &Asm[arow * APAD + acol];
            *(v8h*)dst       = u.v[0];
            *(v8h*)(dst + 8) = u.v[1];
        }
        // ---- stage B tile transposed (convert f32 -> f16) ----
        {
            const float4* gb = (const float4*)(Bw + (size_t)(k0 + bk) * D_MODEL
                                               + n0 + bn);
#pragma unroll
            for (int i = 0; i < 4; ++i) {
                float4 f = gb[i];
                Bsm[(bn + 4*i + 0) * BPAD + bk] = (_Float16)f.x;
                Bsm[(bn + 4*i + 1) * BPAD + bk] = (_Float16)f.y;
                Bsm[(bn + 4*i + 2) * BPAD + bk] = (_Float16)f.z;
                Bsm[(bn + 4*i + 3) * BPAD + bk] = (_Float16)f.w;
            }
        }
        __syncthreads();

        // ---- compute ----
        v16h afr[2], bfr[4];
#pragma unroll
        for (int i = 0; i < 2; ++i) {
            const _Float16* p = &Asm[(wm*32 + i*16 + ln) * APAD + hw*8];
            afr[i] = ldfrag(p, 16);
        }
#pragma unroll
        for (int j = 0; j < 4; ++j) {
            const _Float16* p = &Bsm[(wn*64 + j*16 + ln) * BPAD + hw*8];
            bfr[j] = ldfrag(p, 16);
        }
#pragma unroll
        for (int i = 0; i < 2; ++i)
#pragma unroll
            for (int j = 0; j < 4; ++j)
                acc[i][j] = wmma16(afr[i], bfr[j], acc[i][j]);
        __syncthreads();
    }

    // ---- epilogue ----
    float bch[4];
#pragma unroll
    for (int j = 0; j < 4; ++j) bch[j] = bias[n0 + wn*64 + j*16 + ln];

#pragma unroll
    for (int i = 0; i < 2; ++i) {
#pragma unroll
        for (int j = 0; j < 4; ++j) {
            const int cc = n0 + wn*64 + j*16 + ln;
#pragma unroll
            for (int r = 0; r < 8; ++r) {
                const int rr = m0 + wm*32 + i*16 + r + hw*8;
                float v = acc[i][j][r] + bch[j];
                if (MODE == 0) {
                    ((float*)outp)[(size_t)rr * D_MODEL + cc] = v;
                } else {
                    v *= scale;
                    const int b  = rr >> 11, s  = rr & (SEQ - 1);
                    const int hh = cc >> 6,  hd = cc & (HEAD_DIM - 1);
                    if (MODE == 1) {
                        ((_Float16*)outp)[(((size_t)(b * N_HEADS + hh)) * SEQ + s)
                                          * HEAD_DIM + hd] = (_Float16)v;
                    } else {  // MODE 2: transposed per-head [BH][HD][S]
                        ((_Float16*)outp)[(((size_t)(b * N_HEADS + hh)) * HEAD_DIM
                                          + hd) * SEQ + s] = (_Float16)v;
                    }
                }
            }
        }
    }
}

// ---------------- Flash attention ----------------
// Q,K: f16 [BH=64][S][64] (Q pre-scaled by 1/sqrt(64)); Vt: f16 [BH][64][S].
// O: f32 [B][S][D_MODEL].
// Block: 256 thr (8 waves), BQ=128 (16 q-rows per wave), kv tiles of 64.
// K and V^T tiles are DMA'd global->LDS via the ASYNCcnt path (no VGPR bounce).
// Causal: kv tiles past the block's last q row are skipped entirely.

#define BQ    128
#define BKV   64
#define KPAD  72   // halves per LDS row (144B: 16B-aligned, bank-spread)

__global__ __launch_bounds__(256)
void mha_attn_kernel(const _Float16* __restrict__ Q, const _Float16* __restrict__ Kt,
                     const _Float16* __restrict__ Vt, float* __restrict__ O)
{
    __shared__ _Float16 Kls[BKV * KPAD];            // [kv][hd]
    __shared__ _Float16 Vls[HEAD_DIM * KPAD];       // [hd][kv]
    __shared__ _Float16 Pls[8 * 16 * KPAD];         // per-wave 16 x 64 P tile

    const int tid  = threadIdx.x;
    const int lane = tid & 31;
    const int w    = tid >> 5;
    const int hw   = lane >> 4;
    const int ln   = lane & 15;
    const int bh   = blockIdx.y;                    // 0..63 (= b*16 + head)
    const int qs   = blockIdx.x * BQ;
    const size_t base  = (size_t)bh * SEQ * HEAD_DIM;   // Q/K per-head base
    const size_t vbase = (size_t)bh * HEAD_DIM * SEQ;   // V^T per-head base

    // Q fragments for this wave's 16 rows (kept in registers)
    v16h qf[2];
    {
        const _Float16* qp = Q + base + (size_t)(qs + w*16 + ln) * HEAD_DIM;
#pragma unroll
        for (int kk = 0; kk < 2; ++kk)
            qf[kk] = ldfrag(qp + kk*32 + hw*8, 16);
    }

    v8f o[4];
#pragma unroll
    for (int j = 0; j < 4; ++j)
#pragma unroll
        for (int r = 0; r < 8; ++r) o[j][r] = 0.0f;
    float mrun[8], lrun[8];
#pragma unroll
    for (int r = 0; r < 8; ++r) { mrun[r] = -INFINITY; lrun[r] = 0.0f; }

    const int kv_max = qs + BQ;                     // causal upper bound (excl)
    for (int kv0 = 0; kv0 < kv_max; kv0 += BKV) {
        // ---- stage K [kv][hd] and V^T [hd][kv] tiles: straight copies ----
        {
#pragma unroll
            for (int i = 0; i < 2; ++i) {
                const int c   = tid * 2 + i;        // 0..511 chunk id (16B each)
                const int row = c >> 3;             // 0..63 (kv for K, hd for V)
                const int prt = (c & 7) * 8;        // half offset 0..56
                const _Float16* kg = Kt + base + (size_t)(kv0 + row) * HEAD_DIM + prt;
                _Float16*       kl = &Kls[row * KPAD + prt];
                const _Float16* vg = Vt + vbase + (size_t)row * SEQ + kv0 + prt;
                _Float16*       vl = &Vls[row * KPAD + prt];
#if HAVE_ASYNC_LDS
                async_copy16(kg, kl);
                async_copy16(vg, vl);
#else
                *(v8h*)kl = *(const v8h*)kg;
                *(v8h*)vl = *(const v8h*)vg;
#endif
            }
#if HAVE_ASYNC_LDS
            wait_async();
#endif
        }
        __syncthreads();

        // ---- scores S = Q @ K^T (already scaled) ----
        v8f st[4];
#pragma unroll
        for (int j = 0; j < 4; ++j) {
#pragma unroll
            for (int r = 0; r < 8; ++r) st[j][r] = 0.0f;
#pragma unroll
            for (int kk = 0; kk < 2; ++kk) {
                v16h bf = ldfrag(&Kls[(j*16 + ln) * KPAD + kk*32 + hw*8], 16);
                st[j] = wmma16(qf[kk], bf, st[j]);
            }
        }

        // ---- causal mask + online softmax (rows live in one 16-lane half) ----
        const int qrow0 = qs + w*16 + hw*8;
        float mloc[8];
#pragma unroll
        for (int r = 0; r < 8; ++r) mloc[r] = -INFINITY;
#pragma unroll
        for (int j = 0; j < 4; ++j) {
            const int kvg = kv0 + j*16 + ln;
#pragma unroll
            for (int r = 0; r < 8; ++r) {
                float sv = st[j][r];
                if (kvg > qrow0 + r) sv = -INFINITY;
                st[j][r] = sv;
                mloc[r] = fmaxf(mloc[r], sv);
            }
        }
#pragma unroll
        for (int r = 0; r < 8; ++r) {
#pragma unroll
            for (int msk = 1; msk < 16; msk <<= 1)
                mloc[r] = fmaxf(mloc[r], __shfl_xor(mloc[r], msk, 16));
        }
        float alpha[8], lsum[8];
#pragma unroll
        for (int r = 0; r < 8; ++r) {
            const float mn = fmaxf(mrun[r], mloc[r]);
            alpha[r] = __expf(mrun[r] - mn);        // 0 on first tile (-inf - fin)
            mrun[r]  = mn;
            lsum[r]  = 0.0f;
        }
#pragma unroll
        for (int j = 0; j < 4; ++j)
#pragma unroll
            for (int r = 0; r < 8; ++r) {
                const float p = __expf(st[j][r] - mrun[r]);   // masked -> 0
                st[j][r] = p;
                lsum[r] += p;
            }
#pragma unroll
        for (int r = 0; r < 8; ++r) {
#pragma unroll
            for (int msk = 1; msk < 16; msk <<= 1)
                lsum[r] += __shfl_xor(lsum[r], msk, 16);
            lrun[r] = lrun[r] * alpha[r] + lsum[r];
        }
#pragma unroll
        for (int j = 0; j < 4; ++j)
#pragma unroll
            for (int r = 0; r < 8; ++r) o[j][r] *= alpha[r];

        // ---- P: C-layout -> A-layout via wave-private LDS bounce ----
        _Float16* pw = &Pls[w * 16 * KPAD];
#pragma unroll
        for (int j = 0; j < 4; ++j)
#pragma unroll
            for (int r = 0; r < 8; ++r)
                pw[(r + hw*8) * KPAD + j*16 + ln] = (_Float16)st[j][r];

        // ---- O += P @ V ----
#pragma unroll
        for (int j = 0; j < 4; ++j) {
#pragma unroll
            for (int kk = 0; kk < 2; ++kk) {
                v16h af = ldfrag(&pw[ln * KPAD + kk*32 + hw*8], 16);
                v16h bf = ldfrag(&Vls[(j*16 + ln) * KPAD + kk*32 + hw*8], 16);
                o[j] = wmma16(af, bf, o[j]);
            }
        }
        __syncthreads();
    }

    // ---- normalize and write [B,S,D_MODEL] f32 ----
    const int b = bh >> 4, hh = bh & 15;
    float inv[8];
#pragma unroll
    for (int r = 0; r < 8; ++r) inv[r] = 1.0f / lrun[r];
#pragma unroll
    for (int j = 0; j < 4; ++j)
#pragma unroll
        for (int r = 0; r < 8; ++r) {
            const int s = qs + w*16 + r + hw*8;
            O[((size_t)b * SEQ + s) * D_MODEL + hh * HEAD_DIM + j*16 + ln]
                = o[j][r] * inv[r];
        }
}

// ---------------- launcher ----------------

extern "C" void kernel_launch(void* const* d_in, const int* in_sizes, int n_in,
                              void* d_out, int out_size, void* d_ws, size_t ws_size,
                              hipStream_t stream) {
    (void)in_sizes; (void)n_in; (void)out_size; (void)ws_size;
    const float* query = (const float*)d_in[0];
    const float* key   = (const float*)d_in[1];
    const float* value = (const float*)d_in[2];
    // d_in[3] = mask: causal tril, computed analytically in-kernel
    const float* Wq = (const float*)d_in[4];
    const float* bq = (const float*)d_in[5];
    const float* Wk = (const float*)d_in[6];
    const float* bk = (const float*)d_in[7];
    const float* Wv = (const float*)d_in[8];
    const float* bv = (const float*)d_in[9];
    const float* Wo = (const float*)d_in[10];
    const float* bo = (const float*)d_in[11];

    char* ws = (char*)d_ws;
    _Float16* qf = (_Float16*)(ws);                               // 16 MB
    _Float16* kf = (_Float16*)(ws + (size_t)16 * 1024 * 1024);    // 16 MB
    _Float16* vt = (_Float16*)(ws + (size_t)32 * 1024 * 1024);    // 16 MB (V^T)
    float*    of = (float*)   (ws + (size_t)48 * 1024 * 1024);    // 32 MB

    dim3 ggrid(D_MODEL / GBN, M_ROWS / GBM);   // (8, 64)
    mha_gemm_kernel<1><<<ggrid, 256, 0, stream>>>(query, Wq, bq, (void*)qf, 0.125f);
    mha_gemm_kernel<1><<<ggrid, 256, 0, stream>>>(key,   Wk, bk, (void*)kf, 1.0f);
    mha_gemm_kernel<2><<<ggrid, 256, 0, stream>>>(value, Wv, bv, (void*)vt, 1.0f);

    dim3 agrid(SEQ / BQ, BATCH * N_HEADS);     // (16, 64)
    mha_attn_kernel<<<agrid, 256, 0, stream>>>(qf, kf, vt, of);

    mha_gemm_kernel<0><<<ggrid, 256, 0, stream>>>(of, Wo, bo, d_out, 1.0f);
}
